// DeformableConvLSTMCell_33895881900284
// MI455X (gfx1250) — matrix operand order repaired and checked by reading, not compiled, not run on hardware
//
#include <hip/hip_runtime.h>
#include <math.h>

#define BQ   64
#define HH   28
#define WW   28
#define CINq 256
#define DDq  256
#define MPIX (BQ * HH * WW)      // 50176
#define KDIM (9 * CINq)          // 2304
#define NELEM (MPIX * DDq)       // 12845056

typedef __attribute__((ext_vector_type(16))) __bf16 v16bf;
typedef __attribute__((ext_vector_type(8)))  float  v8f;

__device__ __forceinline__ unsigned short f2bf(float f) {
  unsigned int u = __float_as_uint(f);
  u += 0x7FFFu + ((u >> 16) & 1u);          // round-to-nearest-even
  return (unsigned short)(u >> 16);
}

// Async global->LDS copy of 16 bytes per lane (CDNA5 ASYNCcnt path, bypasses VGPRs).
__device__ __forceinline__ void async_copy16(unsigned ldsByteAddr, const void* gsrc) {
  asm volatile("global_load_async_to_lds_b128 %0, %1, off"
               :: "v"(ldsByteAddr), "v"((unsigned long long)gsrc)
               : "memory");
}
__device__ __forceinline__ void wait_async_le2() {
  asm volatile("s_wait_asynccnt 0x2" ::: "memory");
}
__device__ __forceinline__ void wait_async_le0() {
  asm volatile("s_wait_asynccnt 0x0" ::: "memory");
}

// ---------------- fp32 -> bf16 conversion ----------------
__global__ void cvt_bf16_kernel(const float* __restrict__ src,
                                unsigned short* __restrict__ dst, int n) {
  int i = blockIdx.x * blockDim.x + threadIdx.x;
  if (i < n) dst[i] = f2bf(src[i]);
}

// w: [KH,KW,CIN,COUT] fp32  ->  bp[cout][k] bf16, k = (kh*3+kw)*CIN + cin
__global__ void pack_weights_kernel(const float* __restrict__ w,
                                    unsigned short* __restrict__ bp) {
  int i = blockIdx.x * blockDim.x + threadIdx.x;   // i = n*KDIM + k
  if (i >= DDq * KDIM) return;
  int n = i / KDIM;
  int k = i - n * KDIM;
  bp[i] = f2bf(w[k * DDq + n]);
}

// zero pad source for SAME-padding OOB lanes (re-written every launch: deterministic)
__global__ void zero_pad_kernel(float* __restrict__ p) { p[threadIdx.x] = 0.0f; }

// ---------------- implicit-GEMM 3x3 SAME conv via bf16 WMMA ----------------
// img: [MPIX][CIN] bf16, bp: [DD][KDIM] bf16, zpad: >=256B of zeros, out: [MPIX][DD] f32
__launch_bounds__(256)
__global__ void conv_wmma_kernel(const unsigned short* __restrict__ img,
                                 const unsigned short* __restrict__ bp,
                                 const unsigned short* __restrict__ zpad,
                                 float* __restrict__ out) {
  __shared__ __align__(16) unsigned short ldsA[2][128 * 32];   // double-buffered im2col tile
  __shared__ __align__(16) unsigned short ldsB[2][128 * 32];   // double-buffered weight tile

  const int t     = threadIdx.x;
  const int lane  = t & 31;
  const int wave  = t >> 5;          // 8 waves
  const int wm    = wave & 3;        // 4 waves along M: 32 rows each
  const int wn    = wave >> 2;       // 2 waves along N: 64 cols each
  const int khalf = lane >> 4;
  const int l16   = lane & 15;

  // LDS fill assignment: 256 threads -> 128 rows x 2 halves (16 bf16 = 32B each)
  const int row  = t >> 1;
  const int half = t & 1;

  // A-tile row -> pixel coordinates (M = 50176 is an exact multiple of 128)
  const int gm   = blockIdx.x * 128 + row;
  const int pb   = gm / (HH * WW);
  const int prem = gm - pb * (HH * WW);
  const int py   = prem / WW;
  const int px   = prem - py * WW;
  // B-tile row -> output channel
  const int gn   = blockIdx.y * 128 + row;

  // LDS byte addresses for this thread's 16B destination chunk
  // (flat address low 32 bits == LDS byte address per ISA 10.2)
  const unsigned ldsAbase = (unsigned)(unsigned long long)(&ldsA[0][0]);
  const unsigned ldsBbase = (unsigned)(unsigned long long)(&ldsB[0][0]);
  const unsigned chunkOff = (unsigned)(row * 32 + half * 16) * 2u;
  const unsigned bufStride = 128u * 32u * 2u;   // 8KB

  v8f acc[2][4] = {};

  // issue async loads for K-step kt into buffer `buf` (exactly 2 per lane -> 2/wave)
  auto issue_tile = [&](int kt, int buf) {
    const int kk = kt * 32;
    const int kh = kk / (3 * CINq);
    const int r2 = kk - kh * (3 * CINq);
    const int kw = r2 >> 8;
    const int ci = (r2 & (CINq - 1)) + half * 16;
    const int iy = py + kh - 1, ix = px + kw - 1;
    const bool inb = (iy >= 0 && iy < HH && ix >= 0 && ix < WW);
    const unsigned short* srcA =
        inb ? (img + (((pb * HH + iy) * WW + ix) << 8) + ci) : zpad;
    async_copy16(ldsAbase + buf * bufStride + chunkOff, srcA);
    const unsigned short* srcB = bp + gn * KDIM + kk + half * 16;
    async_copy16(ldsBbase + buf * bufStride + chunkOff, srcB);
  };

  issue_tile(0, 0);   // prologue

  for (int kt = 0; kt < KDIM / 32; ++kt) {
    const int cur = kt & 1;
    if (kt + 1 < KDIM / 32) {
      issue_tile(kt + 1, cur ^ 1);  // overlap: next tile streams while we compute
      wait_async_le2();             // FIFO => tile kt's 2 loads have completed
    } else {
      wait_async_le0();
    }
    __syncthreads();                // all waves' async writes for tile kt visible

    const unsigned int* lda32 = (const unsigned int*)&ldsA[cur][0];
    const unsigned int* ldb32 = (const unsigned int*)&ldsB[cur][0];

    // ---- fragment loads per documented CDNA5 bf16 layouts ----
    union Frag { unsigned int u[8]; v16bf v; };
    Frag af[2], bfr[4];
#pragma unroll
    for (int mi = 0; mi < 2; ++mi) {
      const int mrow = wm * 32 + mi * 16 + l16;
#pragma unroll
      for (int j = 0; j < 8; ++j) {
        // A 16x32: lanes 0-15 K{0-7,16-23}, lanes 16-31 K{8-15,24-31}, pairs packed
        const int kidx = (j < 4) ? (khalf * 4 + j) : (8 + khalf * 4 + (j - 4));
        af[mi].u[j] = lda32[mrow * 16 + kidx];
      }
    }
#pragma unroll
    for (int ni = 0; ni < 4; ++ni) {
      const int nrow = wn * 64 + ni * 16 + l16;
#pragma unroll
      for (int j = 0; j < 8; ++j)  // B 32x16: lanes 0-15 K{0-7,16-23}, lanes 16-31 K{8-15,24-31}
        bfr[ni].u[j] = ldb32[nrow * 16 + ((j < 4) ? (khalf * 4 + j) : (8 + khalf * 4 + (j - 4)))];
    }

#pragma unroll
    for (int mi = 0; mi < 2; ++mi)
#pragma unroll
      for (int ni = 0; ni < 4; ++ni)
        acc[mi][ni] = __builtin_amdgcn_wmma_f32_16x16x32_bf16(
            false, af[mi].v, false, bfr[ni].v,
            (short)0, acc[mi][ni], false, false);

    __syncthreads();   // fence buf[cur] reads before tile kt+2 overwrites it
  }

  // ---- epilogue: C/D layout VGPR r -> M = r + 8*khalf, N = lane%16 ----
#pragma unroll
  for (int mi = 0; mi < 2; ++mi) {
#pragma unroll
    for (int ni = 0; ni < 4; ++ni) {
      const int Ng = blockIdx.y * 128 + wn * 64 + ni * 16 + l16;
#pragma unroll
      for (int r = 0; r < 8; ++r) {
        const int Mg = blockIdx.x * 128 + wm * 32 + mi * 16 + khalf * 8 + r;
        out[Mg * DDq + Ng] = acc[mi][ni][r];
      }
    }
  }
}

// ---------------- GAP + sigmoid gates ----------------
// gates layout: [3][BQ][DD]  (i, f, o)
__global__ void gates_kernel(const float* __restrict__ convX,
                             const float* __restrict__ hidden,
                             const float* __restrict__ wxi, const float* __restrict__ whi,
                             const float* __restrict__ bi,
                             const float* __restrict__ wxf, const float* __restrict__ whf,
                             const float* __restrict__ bfg,
                             const float* __restrict__ wxo, const float* __restrict__ who,
                             const float* __restrict__ bo,
                             float* __restrict__ gates) {
  const int b = blockIdx.x;
  const int d = threadIdx.x;
  const float* pX = convX  + (size_t)b * HH * WW * DDq + d;
  const float* pH = hidden + (size_t)b * HH * WW * DDq + d;
  float si = 0.f, sh = 0.f;
  for (int p = 0; p < HH * WW; ++p) {
    si += pX[p * DDq];
    sh += pH[p * DDq];
  }
  const float gi = si * (1.0f / (HH * WW));
  const float gh = sh * (1.0f / (HH * WW));
  const int o = b * DDq + d;
  gates[o]                 = 1.f / (1.f + __expf(-(wxi[d] * gi + whi[d] * gh + bi[d])));
  gates[BQ * DDq + o]      = 1.f / (1.f + __expf(-(wxf[d] * gi + whf[d] * gh + bfg[d])));
  gates[2 * BQ * DDq + o]  = 1.f / (1.f + __expf(-(wxo[d] * gi + who[d] * gh + bo[d])));
}

// ---------------- fused LSTM elementwise ----------------
__global__ void fuse_kernel(const float* __restrict__ convX,
                            const float* __restrict__ convH,
                            const float* __restrict__ gateBias,   // [HH*WW][DD]
                            const float* __restrict__ statePrev,
                            const float* __restrict__ gates,      // [3][BQ][DD]
                            float* __restrict__ out) {            // hidden | state
  int idx = blockIdx.x * blockDim.x + threadIdx.x;
  if (idx >= NELEM) return;
  const int d   = idx & (DDq - 1);
  const int pix = idx >> 8;
  const int b   = pix / (HH * WW);
  const int sp  = pix - b * (HH * WW);
  const float g = tanhf(convX[idx] + convH[idx] + gateBias[sp * DDq + d]);
  const int go  = b * DDq + d;
  const float fi = gates[go];
  const float ff = gates[BQ * DDq + go];
  const float fo = gates[2 * BQ * DDq + go];
  const float st = ff * statePrev[idx] + fi * g;
  out[idx]         = fo * tanhf(st);
  out[NELEM + idx] = st;
}

extern "C" void kernel_launch(void* const* d_in, const int* in_sizes, int n_in,
                              void* d_out, int out_size, void* d_ws, size_t ws_size,
                              hipStream_t stream) {
  const float* inputs      = (const float*)d_in[0];
  const float* state_prev  = (const float*)d_in[1];
  const float* hidden_prev = (const float*)d_in[2];
  const float* wconvInput  = (const float*)d_in[3];
  const float* wconvHidden = (const float*)d_in[4];
  const float* gateBias    = (const float*)d_in[5];
  const float* wxi = (const float*)d_in[6];
  const float* whi = (const float*)d_in[7];
  const float* bi  = (const float*)d_in[8];
  const float* wxf = (const float*)d_in[9];
  const float* whf = (const float*)d_in[10];
  const float* bfg = (const float*)d_in[11];
  const float* wxo = (const float*)d_in[12];
  const float* who = (const float*)d_in[13];
  const float* bo  = (const float*)d_in[14];
  float* out = (float*)d_out;

  // workspace carve-out (~156 MB total)
  char* ws = (char*)d_ws;
  size_t off = 0;
  auto take = [&](size_t bytes) -> char* {
    char* p = ws + off;
    off = (off + bytes + 255) & ~(size_t)255;
    return p;
  };
  unsigned short* Ax = (unsigned short*)take((size_t)NELEM * 2);
  unsigned short* Ah = (unsigned short*)take((size_t)NELEM * 2);
  unsigned short* Bx = (unsigned short*)take((size_t)DDq * KDIM * 2);
  unsigned short* Bh = (unsigned short*)take((size_t)DDq * KDIM * 2);
  float* convX = (float*)take((size_t)NELEM * 4);
  float* convH = (float*)take((size_t)NELEM * 4);
  float* gates = (float*)take((size_t)3 * BQ * DDq * 4);
  float* zpad  = (float*)take(256);
  (void)ws_size; (void)in_sizes; (void)n_in; (void)out_size;

  zero_pad_kernel<<<1, 64, 0, stream>>>(zpad);
  cvt_bf16_kernel<<<(NELEM + 255) / 256, 256, 0, stream>>>(inputs, Ax, NELEM);
  cvt_bf16_kernel<<<(NELEM + 255) / 256, 256, 0, stream>>>(hidden_prev, Ah, NELEM);
  pack_weights_kernel<<<(DDq * KDIM + 255) / 256, 256, 0, stream>>>(wconvInput, Bx);
  pack_weights_kernel<<<(DDq * KDIM + 255) / 256, 256, 0, stream>>>(wconvHidden, Bh);

  dim3 gemmGrid(MPIX / 128, DDq / 128);
  conv_wmma_kernel<<<gemmGrid, 256, 0, stream>>>(Ax, Bx, (const unsigned short*)zpad, convX);
  conv_wmma_kernel<<<gemmGrid, 256, 0, stream>>>(Ah, Bh, (const unsigned short*)zpad, convH);

  gates_kernel<<<BQ, DDq, 0, stream>>>(convX, hidden_prev, wxi, whi, bi,
                                       wxf, whf, bfg, wxo, who, bo, gates);
  fuse_kernel<<<(NELEM + 255) / 256, 256, 0, stream>>>(convX, convH, gateBias,
                                                       state_prev, gates, out);
}
